// SpikingNeuralNetwork_65214783422962
// MI455X (gfx1250) — matrix operand order repaired and checked by reading, not compiled, not run on hardware
//
#include <hip/hip_runtime.h>
#include <hip/hip_bf16.h>

typedef __attribute__((ext_vector_type(16))) _Float16 v16h;
typedef __attribute__((ext_vector_type(8)))  _Float16 v8h;
typedef __attribute__((ext_vector_type(8)))  float    v8f;

union HalfVec16 {
    v16h v;
    v8h  h8[2];
};

// ---------------------------------------------------------------------------
// Encoder: A[b,f] = mean_t( (x[b,t,f] - x[b-1,t,f]) >= 0.2 ), b==0 -> 0.
// x: [256,128,512] f32 ; A: [256,512] f16  (values k/128, exact in f16)
// ---------------------------------------------------------------------------
__global__ void snn_delta_mean(const float* __restrict__ x,
                               _Float16* __restrict__ A,
                               int B, int T, int F) {
    int f = blockIdx.x * blockDim.x + threadIdx.x;
    int b = blockIdx.y;
    if (f >= F) return;
    int bp = (b > 0) ? (b - 1) : 0;
    const float* xb = x + (long)b  * T * F + f;
    const float* xp = x + (long)bp * T * F + f;
    float cnt = 0.0f;
    for (int t = 0; t < T; ++t) {
        float d = xb[(long)t * F] - xp[(long)t * F];
        cnt += (d >= 0.2f) ? 1.0f : 0.0f;
    }
    A[(long)b * F + f] = (_Float16)(cnt * (1.0f / 128.0f));
}

// ---------------------------------------------------------------------------
// Weight conversion f32 -> f16
// ---------------------------------------------------------------------------
__global__ void snn_f32_to_f16(const float* __restrict__ in,
                               _Float16* __restrict__ out, long n) {
    long i = (long)blockIdx.x * blockDim.x + threadIdx.x;
    if (i < n) out[i] = (_Float16)in[i];
}

// ---------------------------------------------------------------------------
// Batched GEMM with bias:  C[s] = A[s] (MxK, f16) * B^T (B is NxK, f16) + bias
// One wave computes a 32(M) x 64(N) tile: 2 A fragments x 4 B fragments per
// K=32 step -> 8 x v_wmma_f32_16x16x32_f16 per 12 b128 loads (1.5 loads/WMMA).
// Block = 256 threads = 8 waves stacked along M (block tile 256 x 64).
// A/B fragments: contiguous k = 16*lhalf + j in both; C/D layout per CDNA5 ISA 7.12.2.
// ---------------------------------------------------------------------------
__global__ void snn_wmma_gemm_bias(const _Float16* __restrict__ Aall,
                                   const _Float16* __restrict__ Bmat,
                                   const float* __restrict__ bias,
                                   float* __restrict__ Call,
                                   int M, int N, int K,
                                   long aBatchStride, long cBatchStride) {
    const int s    = blockIdx.z;
    const _Float16* A = Aall + (long)s * aBatchStride;
    float*          C = Call + (long)s * cBatchStride;

    const int wave  = threadIdx.x >> 5;   // 0..7
    const int lane  = threadIdx.x & 31;
    const int l16   = lane & 15;
    const int lhalf = lane >> 4;          // 0 or 1

    const int m0 = blockIdx.y * 256 + wave * 32;
    const int n0 = blockIdx.x * 64;
    if (m0 >= M) return;                  // uniform per wave

    v8f acc00 = {}, acc01 = {}, acc02 = {}, acc03 = {};
    v8f acc10 = {}, acc11 = {}, acc12 = {}, acc13 = {};

    // A fragment addressing: lane l holds row m + l16;
    // halves j=0..15 -> K = k0 + lhalf*16 + j  (one contiguous 32B run)
    const _Float16* aBase0 = A + (long)(m0 + l16) * K + lhalf * 16;
    const _Float16* aBase1 = aBase0 + (long)16 * K;

    // B fragment addressing: lane l holds column n + l16;
    // halves j=0..15 -> K = k0 + lhalf*16 + j  (one contiguous 32B run)
    const _Float16* bBase0 = Bmat + (long)(n0 +  0 + l16) * K + lhalf * 16;
    const _Float16* bBase1 = Bmat + (long)(n0 + 16 + l16) * K + lhalf * 16;
    const _Float16* bBase2 = Bmat + (long)(n0 + 32 + l16) * K + lhalf * 16;
    const _Float16* bBase3 = Bmat + (long)(n0 + 48 + l16) * K + lhalf * 16;

    for (int k0 = 0; k0 < K; k0 += 32) {
        HalfVec16 a0, a1;
        a0.h8[0] = *(const v8h*)(aBase0 + k0);
        a0.h8[1] = *(const v8h*)(aBase0 + k0 + 8);
        a1.h8[0] = *(const v8h*)(aBase1 + k0);
        a1.h8[1] = *(const v8h*)(aBase1 + k0 + 8);

        HalfVec16 b0, b1, b2, b3;
        b0.h8[0] = *(const v8h*)(bBase0 + k0);
        b0.h8[1] = *(const v8h*)(bBase0 + k0 + 8);
        b1.h8[0] = *(const v8h*)(bBase1 + k0);
        b1.h8[1] = *(const v8h*)(bBase1 + k0 + 8);
        b2.h8[0] = *(const v8h*)(bBase2 + k0);
        b2.h8[1] = *(const v8h*)(bBase2 + k0 + 8);
        b3.h8[0] = *(const v8h*)(bBase3 + k0);
        b3.h8[1] = *(const v8h*)(bBase3 + k0 + 8);

        acc00 = __builtin_amdgcn_wmma_f32_16x16x32_f16(false, a0.v, false, b0.v,
                                                       (short)0, acc00, false, false);
        acc01 = __builtin_amdgcn_wmma_f32_16x16x32_f16(false, a0.v, false, b1.v,
                                                       (short)0, acc01, false, false);
        acc02 = __builtin_amdgcn_wmma_f32_16x16x32_f16(false, a0.v, false, b2.v,
                                                       (short)0, acc02, false, false);
        acc03 = __builtin_amdgcn_wmma_f32_16x16x32_f16(false, a0.v, false, b3.v,
                                                       (short)0, acc03, false, false);
        acc10 = __builtin_amdgcn_wmma_f32_16x16x32_f16(false, a1.v, false, b0.v,
                                                       (short)0, acc10, false, false);
        acc11 = __builtin_amdgcn_wmma_f32_16x16x32_f16(false, a1.v, false, b1.v,
                                                       (short)0, acc11, false, false);
        acc12 = __builtin_amdgcn_wmma_f32_16x16x32_f16(false, a1.v, false, b2.v,
                                                       (short)0, acc12, false, false);
        acc13 = __builtin_amdgcn_wmma_f32_16x16x32_f16(false, a1.v, false, b3.v,
                                                       (short)0, acc13, false, false);
    }

    // Epilogue. C/D layout: VGPR r holds row mBlock + lhalf*8 + r, col n + l16.
    v8f accA[4] = {acc00, acc01, acc02, acc03};
    v8f accB[4] = {acc10, acc11, acc12, acc13};
#pragma unroll
    for (int i = 0; i < 4; ++i) {
        const int col = n0 + i * 16 + l16;
        const float bv = bias[col];
#pragma unroll
        for (int r = 0; r < 8; ++r) {
            const int row0 = m0 + lhalf * 8 + r;
            C[(long)row0 * N + col]        = accA[i][r] + bv;
            C[(long)(row0 + 16) * N + col] = accB[i][r] + bv;
        }
    }
}

// ---------------------------------------------------------------------------
// LIF scan (snntorch Leaky, reset_mechanism='subtract', beta=0.9, thr=1):
//   reset = (mem > 1); mem = 0.9*mem + cur - reset; spk = (mem > 1)
// stepStride==0 handles the layer-0 case (same current every step).
// ---------------------------------------------------------------------------
__global__ void snn_lif_to_f16(const float* __restrict__ cur, long stepStride,
                               _Float16* __restrict__ spk, long n) {
    long i = (long)blockIdx.x * blockDim.x + threadIdx.x;
    if (i >= n) return;
    float mem = 0.0f;
    for (int s = 0; s < 64; ++s) {
        float c = cur[i + (long)s * stepStride];
        float reset = (mem > 1.0f) ? 1.0f : 0.0f;
        mem = 0.9f * mem + c - reset;
        spk[i + (long)s * n] = (_Float16)((mem > 1.0f) ? 1.0f : 0.0f);
    }
}

__global__ void snn_lif_to_f32(const float* __restrict__ cur, long stepStride,
                               float* __restrict__ out, long n) {
    long i = (long)blockIdx.x * blockDim.x + threadIdx.x;
    if (i >= n) return;
    float mem = 0.0f;
    for (int s = 0; s < 64; ++s) {
        float c = cur[i + (long)s * stepStride];
        float reset = (mem > 1.0f) ? 1.0f : 0.0f;
        mem = 0.9f * mem + c - reset;
        out[i + (long)s * n] = (mem > 1.0f) ? 1.0f : 0.0f;
    }
}

// ---------------------------------------------------------------------------
// Host side
// ---------------------------------------------------------------------------
extern "C" void kernel_launch(void* const* d_in, const int* in_sizes, int n_in,
                              void* d_out, int out_size, void* d_ws, size_t ws_size,
                              hipStream_t stream) {
    (void)in_sizes; (void)n_in; (void)out_size; (void)ws_size;

    // Inputs (setup_inputs order): x, W0, b0, W1, b1, W2, b2
    const float* x  = (const float*)d_in[0];   // [256,128,512]
    const float* W0 = (const float*)d_in[1];   // [1024,512]
    const float* b0 = (const float*)d_in[2];   // [1024]
    const float* W1 = (const float*)d_in[3];   // [1024,1024]
    const float* b1 = (const float*)d_in[4];   // [1024]
    const float* W2 = (const float*)d_in[5];   // [512,1024]
    const float* b2 = (const float*)d_in[6];   // [512]
    float* out = (float*)d_out;                // [64,256,512]

    const int B = 256, T = 128, F = 512, H1 = 1024, H2 = 1024, H3 = 512, S = 64;

    // Workspace layout (bytes)
    char* ws = (char*)d_ws;
    _Float16* A16  = (_Float16*)(ws + 0);                 // 256*512*2    = 256 KB
    _Float16* W0h  = (_Float16*)(ws + (1 << 20));         // 1024*512*2   = 1 MB
    _Float16* W1h  = (_Float16*)(ws + (2 << 20));         // 1024*1024*2  = 2 MB
    _Float16* W2h  = (_Float16*)(ws + (4 << 20));         // 512*1024*2   = 1 MB
    float*    cur0 = (float*)   (ws + (5 << 20));         // 256*1024*4   = 1 MB
    _Float16* spk  = (_Float16*)(ws + (6 << 20));         // 64*256*1024*2 = 32 MB (reused L1,L2 in)
    float*    curB = (float*)   (ws + (38l << 20));       // 64*256*1024*4 = 64 MB (reused cur1,cur2)

    // 1) Encoder: delta spikes, mean over T -> f16 A [256,512]
    snn_delta_mean<<<dim3((F + 255) / 256, B, 1), 256, 0, stream>>>(x, A16, B, T, F);

    // 2) Weights -> f16
    snn_f32_to_f16<<<(H1 * F + 255) / 256, 256, 0, stream>>>(W0, W0h, (long)H1 * F);
    snn_f32_to_f16<<<(H2 * H1 + 255) / 256, 256, 0, stream>>>(W1, W1h, (long)H2 * H1);
    snn_f32_to_f16<<<(H3 * H2 + 255) / 256, 256, 0, stream>>>(W2, W2h, (long)H3 * H2);

    // 3) Layer 0: input current identical for all 64 steps -> one GEMM 256x512x1024
    snn_wmma_gemm_bias<<<dim3(H1 / 64, B / 256, 1), 256, 0, stream>>>(
        A16, W0h, b0, cur0, B, H1, F, 0, 0);

    // 4) LIF 0 (broadcast current) -> spk [64,256,1024] f16
    snn_lif_to_f16<<<(B * H1 + 255) / 256, 256, 0, stream>>>(
        cur0, 0l, spk, (long)B * H1);

    // 5) Layer 1: 64 batched GEMMs 256x1024x1024 -> curB
    snn_wmma_gemm_bias<<<dim3(H2 / 64, B / 256, S), 256, 0, stream>>>(
        spk, W1h, b1, curB, B, H2, H1, (long)B * H1, (long)B * H2);

    // 6) LIF 1 -> spk (reuse buffer) [64,256,1024] f16
    snn_lif_to_f16<<<(B * H2 + 255) / 256, 256, 0, stream>>>(
        curB, (long)B * H2, spk, (long)B * H2);

    // 7) Layer 2: 64 batched GEMMs 256x1024x512 -> curB (reuse, fits)
    snn_wmma_gemm_bias<<<dim3(H3 / 64, B / 256, S), 256, 0, stream>>>(
        spk, W2h, b2, curB, B, H3, H2, (long)B * H2, (long)B * H3);

    // 8) LIF 2 -> d_out [64,256,512] f32
    snn_lif_to_f32<<<(B * H3 + 255) / 256, 256, 0, stream>>>(
        curB, (long)B * H3, out, (long)B * H3);
}
